// Model_33672543600675
// MI455X (gfx1250) — matrix operand order repaired and checked by reading, not compiled, not run on hardware
//
#include <hip/hip_runtime.h>

typedef __attribute__((ext_vector_type(16))) __bf16 v16bf;
typedef __attribute__((ext_vector_type(8)))  float  v8f;

#define TOK 8192   // B*S
#define SEQ 2048
#define DM  512
#define NH  8
#define HD  64
#define DFF 2048
#define NV  512

// ---------- helpers ----------
__device__ __forceinline__ unsigned short f2b(float f) {
  union { float f; unsigned int u; } x; x.f = f;
  unsigned int r = x.u + 0x7FFFu + ((x.u >> 16) & 1u);   // RNE
  return (unsigned short)(r >> 16);
}
__device__ __forceinline__ float b2f(unsigned short b) {
  union { unsigned int u; float f; } x; x.u = ((unsigned int)b) << 16;
  return x.f;
}
// build a 16 x bf16 fragment from two 16B chunks
__device__ __forceinline__ v16bf load16(const unsigned short* p0, const unsigned short* p1) {
  union { v16bf v; uint4 q[2]; } u;
  u.q[0] = *(const uint4*)p0;
  u.q[1] = *(const uint4*)p1;
  return u.v;
}
__device__ __forceinline__ v8f wmma_bf16(v16bf a, v16bf b, v8f c) {
  return __builtin_amdgcn_wmma_f32_16x16x32_bf16(false, a, false, b, (short)0, c, false, false);
}
// async global->LDS copy of 16 bytes per lane (CDNA5, tracked by ASYNCcnt)
__device__ __forceinline__ void async_cp16(void* lds, const void* gptr) {
  unsigned           l = (unsigned)(uintptr_t)lds;          // addr[31:0] = LDS offset
  unsigned long long g = (unsigned long long)(uintptr_t)gptr;
  asm volatile("global_load_async_to_lds_b128 %0, %1, off" :: "v"(l), "v"(g) : "memory");
}
__device__ __forceinline__ void wait_async0() {
  asm volatile("s_wait_asynccnt 0x0" ::: "memory");
}

// ---------- embedding + sinusoidal PE ----------
__global__ void embed_k(const int* __restrict__ X, const float* __restrict__ E,
                        float* __restrict__ H) {
  size_t gid = (size_t)blockIdx.x * 256 + threadIdx.x;   // over TOK*DM
  int d = (int)(gid % DM); size_t t = gid / DM;
  int s = (int)(t % SEQ);
  int tok = X[t];
  float v = E[(size_t)tok * DM + d];
  float freq = __expf(-(float)(d & ~1) * (9.210340371976184f / (float)DM));
  float ang = (float)s * freq;
  v += (d & 1) ? __cosf(ang) : __sinf(ang);
  H[gid] = v;
}

// ---------- fp32 -> bf16 convert ----------
__global__ void cvt_k(const float* __restrict__ in, unsigned short* __restrict__ out, int n) {
  int i = blockIdx.x * 256 + threadIdx.x;
  if (i < n) out[i] = f2b(in[i]);
}

// ---------- layernorm (fp32 in, bf16 out), one block per row of 512 ----------
__global__ void ln_k(const float* __restrict__ X, const float* __restrict__ G,
                     const float* __restrict__ Bt, unsigned short* __restrict__ O) {
  __shared__ float red[256];
  int row = blockIdx.x, tid = threadIdx.x;
  const float* x = X + (size_t)row * DM;
  float v0 = x[tid], v1 = x[tid + 256];
  red[tid] = v0 + v1; __syncthreads();
  for (int o = 128; o > 0; o >>= 1) { if (tid < o) red[tid] += red[tid + o]; __syncthreads(); }
  float mu = red[0] * (1.0f / DM); __syncthreads();
  float d0 = v0 - mu, d1 = v1 - mu;
  red[tid] = d0 * d0 + d1 * d1; __syncthreads();
  for (int o = 128; o > 0; o >>= 1) { if (tid < o) red[tid] += red[tid + o]; __syncthreads(); }
  float rstd = rsqrtf(red[0] * (1.0f / DM) + 1e-5f);
  O[(size_t)row * DM + tid]       = f2b(d0 * rstd * G[tid] + Bt[tid]);
  O[(size_t)row * DM + tid + 256] = f2b(d1 * rstd * G[tid + 256] + Bt[tid + 256]);
}

// ---------- WMMA bf16 GEMM with async double-buffered LDS staging ----------
// block = 128 threads (4 waves); block tile 64(M) x 128(N); K-step 32.
// A tile 64x32 and B tile 32x128 staged to LDS via global_load_async_to_lds_b128;
// stage of step i+1 overlaps WMMA math of step i.
__global__ void gemm_bf16_k(const unsigned short* __restrict__ A,
                            const unsigned short* __restrict__ B,
                            const float* __restrict__ bias,
                            float* __restrict__ Cf,
                            unsigned short* __restrict__ Cb,
                            const float* __restrict__ Res,
                            int K, int N, int relu) {
  __shared__ unsigned short Abuf[2][64 * 32];   // 2 x 4 KB
  __shared__ unsigned short Bbuf[2][32 * 128];  // 2 x 8 KB
  int tid  = threadIdx.x;
  int lane = tid & 31;
  int wid  = tid >> 5;
  int m0b = blockIdx.x * 64;
  int n0b = blockIdx.y * 128;
  int mw  = (wid & 1) * 32;     // wave M offset in block tile
  int nw  = (wid >> 1) * 64;    // wave N offset in block tile
  int arow = lane & 15;
  int asel = (lane >> 4) << 3;  // K-half select per A-fragment layout
  int lrow = (lane >> 4) << 3;
  int col  = lane & 15;

  auto stage = [&](unsigned short* ab, unsigned short* bb, int k0) {
#pragma unroll
    for (int i = 0; i < 2; ++i) {             // A: 64x32 bf16 = 256 16B-chunks
      int c = tid + i * 128;
      int row = c >> 2, off = (c & 3) * 8;
      async_cp16(ab + row * 32 + off, A + (size_t)(m0b + row) * K + k0 + off);
    }
#pragma unroll
    for (int i = 0; i < 4; ++i) {             // B: 32x128 bf16 = 512 16B-chunks
      int c = tid + i * 128;
      int row = c >> 4, off = (c & 15) * 8;
      async_cp16(bb + row * 128 + off, B + (size_t)(k0 + row) * N + n0b + off);
    }
  };

  v8f acc[2][4] = {};
  stage(Abuf[0], Bbuf[0], 0);
  int nk = K >> 5;
  for (int ki = 0; ki < nk; ++ki) {
    wait_async0();            // my async copies for buffer ki done
    __syncthreads();          // everyone's copies done
    const unsigned short* ab = Abuf[ki & 1];
    const unsigned short* bb = Bbuf[ki & 1];
    if (ki + 1 < nk) stage(Abuf[(ki + 1) & 1], Bbuf[(ki + 1) & 1], (ki + 1) << 5);
    v16bf af[2];
#pragma unroll
    for (int mi = 0; mi < 2; ++mi) {
      const unsigned short* pa = ab + (size_t)(mw + mi * 16 + arow) * 32 + asel;
      af[mi] = load16(pa, pa + 16);
    }
    v16bf bf4[4];
#pragma unroll
    for (int ni = 0; ni < 4; ++ni) {
      const unsigned short* pb = bb + nw + ni * 16 + col;
      union { v16bf v; unsigned short u[16]; } tb;
#pragma unroll
      for (int s = 0; s < 8; ++s) { tb.u[s] = pb[(size_t)(asel + s) * 128]; tb.u[s + 8] = pb[(size_t)(16 + asel + s) * 128]; }
      bf4[ni] = tb.v;
    }
#pragma unroll
    for (int mi = 0; mi < 2; ++mi)
#pragma unroll
      for (int ni = 0; ni < 4; ++ni)
        acc[mi][ni] = wmma_bf16(af[mi], bf4[ni], acc[mi][ni]);
    __syncthreads();          // all waves done reading buffers before next overwrite
  }

#pragma unroll
  for (int mi = 0; mi < 2; ++mi)
#pragma unroll
    for (int ni = 0; ni < 4; ++ni)
#pragma unroll
      for (int r = 0; r < 8; ++r) {
        int m = m0b + mw + mi * 16 + lrow + r;
        int n = n0b + nw + ni * 16 + col;
        float v = acc[mi][ni][r] + bias[n];
        if (relu) v = fmaxf(v, 0.0f);
        size_t idx = (size_t)m * N + n;
        if (Res) v += Res[idx];
        if (Cf) Cf[idx] = v;
        if (Cb) Cb[idx] = f2b(v);
      }
}

// ---------- transpose K into [b][h][d][s] so QK^T B-fragments are contiguous ----------
__global__ void transk_k(const unsigned short* __restrict__ Kb, unsigned short* __restrict__ KT) {
  size_t gid = (size_t)blockIdx.x * 256 + threadIdx.x;   // over TOK*DM
  int s = (int)(gid % SEQ); size_t r = gid / SEQ;
  int d = (int)(r % HD); int bh = (int)(r / HD);
  int b = bh >> 3, h = bh & 7;
  KT[gid] = Kb[((size_t)b * SEQ + s) * DM + h * HD + d];
}

// ---------- attention: block = 4 waves, 64 query rows of one (b,h); scores strip in LDS ----------
__global__ void attn_k(const unsigned short* __restrict__ Q,
                       const unsigned short* __restrict__ KT,
                       const unsigned short* __restrict__ V,
                       unsigned short* __restrict__ Ctx) {
  const int SROW = SEQ + 8;                     // padded bf16 row stride
  int bh = blockIdx.x;
  int b = bh >> 3, h = bh & 7;
  int wid = threadIdx.x >> 5, lane = threadIdx.x & 31;
  int m0 = blockIdx.y * 64 + wid * 16;          // query row base for this wave
  extern __shared__ char smem[];
  unsigned short* sc = (unsigned short*)smem + (size_t)wid * 16 * SROW;
  float* rsum = (float*)(smem + (size_t)4 * 16 * SROW * 2) + wid * 16;
  int lrow = (lane >> 4) << 3;
  int col  = lane & 15;

  // Q fragments: 16 rows x 64 head-dim = 2 x (16x32)
  v16bf aq[2];
  {
    size_t t = (size_t)b * SEQ + m0 + (lane & 15);
    int dbase = h * HD + ((lane >> 4) << 3);
#pragma unroll
    for (int j = 0; j < 2; ++j) {
      const unsigned short* p = Q + t * DM + dbase + j * 32;
      aq[j] = load16(p, p + 16);
    }
  }

  // pass 1: S = Q K^T * 1/sqrt(64), bf16 strip into LDS
  for (int kt = 0; kt < SEQ; kt += 16) {
    v8f acc = {};
#pragma unroll
    for (int j = 0; j < 2; ++j) {
      const unsigned short* p = KT + ((size_t)bh * HD + j * 32 + lrow) * SEQ + kt + col;
      union { v16bf v; unsigned short u[16]; } tk;
#pragma unroll
      for (int s = 0; s < 8; ++s) { tk.u[s] = p[(size_t)s * SEQ]; tk.u[s + 8] = p[(size_t)(16 + s) * SEQ]; }
      v16bf bk = tk.v;
      acc = wmma_bf16(aq[j], bk, acc);
    }
#pragma unroll
    for (int r = 0; r < 8; ++r)
      sc[(size_t)(lrow + r) * SROW + kt + col] = f2b(acc[r] * 0.125f);
  }
  __syncthreads();

  // pass 2: row softmax (2 lanes per row; exp stored unnormalized, sum saved)
  {
    int row = lane >> 1, half = lane & 1;
    unsigned short* pr = sc + (size_t)row * SROW + half * (SEQ / 2);
    float mx = -3.0e38f;
    for (int i = 0; i < SEQ / 2; i += 8) {
      union { uint4 q; unsigned short s[8]; } u;
      u.q = *(const uint4*)(pr + i);
#pragma unroll
      for (int j = 0; j < 8; ++j) mx = fmaxf(mx, b2f(u.s[j]));
    }
    mx = fmaxf(mx, __shfl_xor(mx, 1));
    float sum = 0.0f;
    for (int i = 0; i < SEQ / 2; i += 8) {
      union { uint4 q; unsigned short s[8]; } u;
      u.q = *(const uint4*)(pr + i);
#pragma unroll
      for (int j = 0; j < 8; ++j) {
        float e = __expf(b2f(u.s[j]) - mx);
        sum += e;
        u.s[j] = f2b(e);
      }
      *(uint4*)(pr + i) = u.q;
    }
    sum += __shfl_xor(sum, 1);
    if (!half) rsum[row] = sum;
  }
  __syncthreads();

  // pass 3: ctx = P @ V (P A-fragments straight from LDS)
  v8f acc4[4] = {};
  for (int k0 = 0; k0 < SEQ; k0 += 32) {
    const unsigned short* pa = sc + (size_t)(lane & 15) * SROW + k0 + ((lane >> 4) << 3);
    v16bf ap = load16(pa, pa + 16);
#pragma unroll
    for (int ni = 0; ni < 4; ++ni) {
      const unsigned short* pv = V + ((size_t)b * SEQ + k0 + lrow) * DM + h * HD + ni * 16 + col;
      union { v16bf v; unsigned short u[16]; } tv;
#pragma unroll
      for (int s = 0; s < 8; ++s) { tv.u[s] = pv[(size_t)s * DM]; tv.u[s + 8] = pv[(size_t)(16 + s) * DM]; }
      v16bf bv = tv.v;
      acc4[ni] = wmma_bf16(ap, bv, acc4[ni]);
    }
  }
#pragma unroll
  for (int ni = 0; ni < 4; ++ni)
#pragma unroll
    for (int r = 0; r < 8; ++r) {
      float inv = 1.0f / rsum[lrow + r];
      size_t t = (size_t)b * SEQ + m0 + lrow + r;
      Ctx[t * DM + h * HD + ni * 16 + col] = f2b(acc4[ni][r] * inv);
    }
}

// ---------- final vocab softmax (512 wide) ----------
__global__ void softmax_k(const float* __restrict__ L, float* __restrict__ P) {
  __shared__ float red[256];
  int row = blockIdx.x, tid = threadIdx.x;
  const float* x = L + (size_t)row * NV;
  float a = x[tid], c = x[tid + 256];
  red[tid] = fmaxf(a, c); __syncthreads();
  for (int o = 128; o > 0; o >>= 1) { if (tid < o) red[tid] = fmaxf(red[tid], red[tid + o]); __syncthreads(); }
  float mx = red[0]; __syncthreads();
  float e0 = __expf(a - mx), e1 = __expf(c - mx);
  red[tid] = e0 + e1; __syncthreads();
  for (int o = 128; o > 0; o >>= 1) { if (tid < o) red[tid] += red[tid + o]; __syncthreads(); }
  float inv = 1.0f / red[0];
  P[(size_t)row * NV + tid]       = e0 * inv;
  P[(size_t)row * NV + tid + 256] = e1 * inv;
}

// ---------- launcher ----------
extern "C" void kernel_launch(void* const* d_in, const int* in_sizes, int n_in,
                              void* d_out, int out_size, void* d_ws, size_t ws_size,
                              hipStream_t stream) {
  (void)in_sizes; (void)n_in; (void)out_size; (void)ws_size;
  const int*   x    = (const int*)  d_in[0];
  const float* emb  = (const float*)d_in[1];
  const float* ln1g = (const float*)d_in[2];
  const float* ln1b = (const float*)d_in[3];
  const float* Wq   = (const float*)d_in[4];
  const float* bq   = (const float*)d_in[5];
  const float* Wk   = (const float*)d_in[6];
  const float* bk   = (const float*)d_in[7];
  const float* Wv   = (const float*)d_in[8];
  const float* bv   = (const float*)d_in[9];
  const float* Wo   = (const float*)d_in[10];
  const float* bo   = (const float*)d_in[11];
  const float* ln2g = (const float*)d_in[12];
  const float* ln2b = (const float*)d_in[13];
  const float* W1   = (const float*)d_in[14];
  const float* b1   = (const float*)d_in[15];
  const float* W2   = (const float*)d_in[16];
  const float* b2   = (const float*)d_in[17];
  const float* Wout = (const float*)d_in[18];
  const float* bout = (const float*)d_in[19];

  char* p = (char*)d_ws;
  auto take = [&](size_t bytes) -> char* {
    char* q = p; p += (bytes + 255) & ~(size_t)255; return q;
  };
  float*          h     = (float*)         take((size_t)TOK * DM * 4);
  unsigned short* h_bf  = (unsigned short*)take((size_t)TOK * DM * 2);
  unsigned short* xn    = (unsigned short*)take((size_t)TOK * DM * 2);
  unsigned short* qb    = (unsigned short*)take((size_t)TOK * DM * 2);
  unsigned short* kb    = (unsigned short*)take((size_t)TOK * DM * 2);
  unsigned short* vb    = (unsigned short*)take((size_t)TOK * DM * 2);
  unsigned short* kT    = (unsigned short*)take((size_t)TOK * DM * 2);
  unsigned short* ctxb  = (unsigned short*)take((size_t)TOK * DM * 2);
  unsigned short* ffb   = (unsigned short*)take((size_t)TOK * DFF * 2);
  unsigned short* WqB   = (unsigned short*)take((size_t)DM * DM * 2);
  unsigned short* WkB   = (unsigned short*)take((size_t)DM * DM * 2);
  unsigned short* WvB   = (unsigned short*)take((size_t)DM * DM * 2);
  unsigned short* WoB   = (unsigned short*)take((size_t)DM * DM * 2);
  unsigned short* W1B   = (unsigned short*)take((size_t)DM * DFF * 2);
  unsigned short* W2B   = (unsigned short*)take((size_t)DFF * DM * 2);
  unsigned short* WoutB = (unsigned short*)take((size_t)DM * NV * 2);

  float* logits = (float*)d_out;
  float* probs  = logits + (size_t)TOK * NV;

  embed_k<<<dim3((TOK * DM) / 256), dim3(256), 0, stream>>>(x, emb, h);

  cvt_k<<<dim3((DM * DM) / 256),  dim3(256), 0, stream>>>(Wq,   WqB,   DM * DM);
  cvt_k<<<dim3((DM * DM) / 256),  dim3(256), 0, stream>>>(Wk,   WkB,   DM * DM);
  cvt_k<<<dim3((DM * DM) / 256),  dim3(256), 0, stream>>>(Wv,   WvB,   DM * DM);
  cvt_k<<<dim3((DM * DM) / 256),  dim3(256), 0, stream>>>(Wo,   WoB,   DM * DM);
  cvt_k<<<dim3((DM * DFF) / 256), dim3(256), 0, stream>>>(W1,   W1B,   DM * DFF);
  cvt_k<<<dim3((DFF * DM) / 256), dim3(256), 0, stream>>>(W2,   W2B,   DFF * DM);
  cvt_k<<<dim3((DM * NV) / 256),  dim3(256), 0, stream>>>(Wout, WoutB, DM * NV);

  auto gemm = [&](const unsigned short* A, const unsigned short* B, const float* bias,
                  float* Cf, unsigned short* Cb, const float* Res, int K, int N, int relu) {
    gemm_bf16_k<<<dim3(TOK / 64, N / 128), dim3(128), 0, stream>>>(A, B, bias, Cf, Cb, Res, K, N, relu);
  };

  ln_k<<<dim3(TOK), dim3(256), 0, stream>>>(h, ln1g, ln1b, xn);
  gemm(xn, WqB, bq, nullptr, qb, nullptr, DM, DM, 0);
  gemm(xn, WkB, bk, nullptr, kb, nullptr, DM, DM, 0);
  gemm(xn, WvB, bv, nullptr, vb, nullptr, DM, DM, 0);

  transk_k<<<dim3((TOK * DM) / 256), dim3(256), 0, stream>>>(kb, kT);

  size_t smem = (size_t)4 * 16 * (SEQ + 8) * 2 + (size_t)4 * 16 * 4;  // ~257 KB of the 320 KB WGP LDS
  attn_k<<<dim3(4 * NH, SEQ / 64), dim3(128), smem, stream>>>(qb, kT, vb, ctxb);

  gemm(ctxb, WoB, bo, h, nullptr, h, DM, DM, 0);                 // attn out + residual
  ln_k<<<dim3(TOK), dim3(256), 0, stream>>>(h, ln2g, ln2b, xn);
  gemm(xn, W1B, b1, nullptr, ffb, nullptr, DM, DFF, 1);          // FFN1 + ReLU
  gemm(ffb, W2B, b2, h, h_bf, h, DFF, DM, 0);                    // FFN2 + residual, dual out
  gemm(h_bf, WoutB, bout, logits, nullptr, nullptr, DM, NV, 0);  // logits
  softmax_k<<<dim3(TOK), dim3(256), 0, stream>>>(logits, probs);
}
